// KalmanNet_1649267442037
// MI455X (gfx1250) — matrix operand order repaired and checked by reading, not compiled, not run on hardware
//
#include <hip/hip_runtime.h>
#include <hip/hip_bf16.h>

typedef __attribute__((ext_vector_type(16))) __bf16 v16bf;
typedef __attribute__((ext_vector_type(8)))  __bf16 v8bf;
typedef __attribute__((ext_vector_type(8)))  float  v8f;

#define MDIM 10
#define NDIM 10
#define HID  2000
#define H1   1600
#define H2   400
#define KHH  2048          // W_hh / W2 K padded to multiple of 32 (and of 1024)
#define NWG  125           // 125 * 16 == HID
#define NTHR 288           // 9 wave32 per WG

// ---------------------------------------------------------------------------
// WMMA helpers (16x16x32 bf16 -> f32). A fragment per documented layout:
// lanes 0-15: VGPR0-3 = K[0..7], VGPR4-7 = K[16..23]; lanes 16-31: +8.
// B (vector replicated over 16 cols): same K layout as A,
// lane half h holds vec[k0 + 8*h .. +8) and vec[k0 + 16 + 8*h .. +8).
// ---------------------------------------------------------------------------
static __device__ __forceinline__ v16bf ld_pair(const __bf16* p0, const __bf16* p1) {
  v8bf lo = *(const v8bf*)p0;
  v8bf hi = *(const v8bf*)p1;
  return __builtin_shufflevector(lo, hi, 0,1,2,3,4,5,6,7,8,9,10,11,12,13,14,15);
}

static __device__ __forceinline__ v8f gemv_tile(const __bf16* __restrict__ Wbase, int ldw,
                                                const __bf16* __restrict__ vec,
                                                int k0, int nchunk, int lane, v8f acc) {
  const int m    = lane & 15;
  const int half = (lane >> 4) & 1;
  const __bf16* wr = Wbase + (size_t)m * ldw + half * 8;
  const __bf16* vp = vec + half * 8;
  for (int c = 0; c < nchunk; ++c) {
    const int k = k0 + (c << 5);
    v16bf a = ld_pair(wr + k, wr + k + 16);
    v16bf b = ld_pair(vp + k, vp + k + 16);
    acc = __builtin_amdgcn_wmma_f32_16x16x32_bf16(false, a, false, b, (short)0, acc,
                                                  false, false);
  }
  return acc;
}

// ---------------------------------------------------------------------------
// Device-wide sense barrier over NWG workgroups (counters live in d_ws).
// ---------------------------------------------------------------------------
static __device__ __forceinline__ void grid_barrier(unsigned* cnt, unsigned* gen) {
  __threadfence();
  __syncthreads();
  if (threadIdx.x == 0) {
    unsigned g = __hip_atomic_load(gen, __ATOMIC_ACQUIRE, __HIP_MEMORY_SCOPE_AGENT);
    unsigned a = __hip_atomic_fetch_add(cnt, 1u, __ATOMIC_ACQ_REL, __HIP_MEMORY_SCOPE_AGENT);
    if (a == (unsigned)(NWG - 1)) {
      __hip_atomic_store(cnt, 0u, __ATOMIC_RELAXED, __HIP_MEMORY_SCOPE_AGENT);
      __hip_atomic_fetch_add(gen, 1u, __ATOMIC_ACQ_REL, __HIP_MEMORY_SCOPE_AGENT);
    } else {
      while (__hip_atomic_load(gen, __ATOMIC_ACQUIRE, __HIP_MEMORY_SCOPE_AGENT) == g)
        __builtin_amdgcn_s_sleep(1);
    }
  }
  __syncthreads();
  __threadfence();
}

// ---------------------------------------------------------------------------
// Prep kernels: fp32 -> bf16 with K-padding; hn / barrier init.
// ---------------------------------------------------------------------------
__global__ void k_cvt(const float* __restrict__ src, __bf16* __restrict__ dst,
                      int rows, int K, int Kp) {
  const int total = rows * Kp;
  for (int j = blockIdx.x * blockDim.x + threadIdx.x; j < total;
       j += gridDim.x * blockDim.x) {
    const int r = j / Kp, c = j - r * Kp;
    dst[j] = (c < K) ? (__bf16)src[(size_t)r * K + c] : (__bf16)0.0f;
  }
}

__global__ void k_init(float* __restrict__ hnf, __bf16* __restrict__ hnb,
                       const float* __restrict__ hn0, unsigned* __restrict__ bar) {
  const int i = blockIdx.x * blockDim.x + threadIdx.x;
  if (i < 2) bar[i] = 0u;
  for (int j = i; j < 2 * KHH; j += gridDim.x * blockDim.x) {
    const int b = j / KHH, idx = j - b * KHH;
    const float v = (b == 0 && idx < HID) ? hn0[idx] : 0.0f;
    hnf[j] = v;
    hnb[j] = (__bf16)v;
  }
}

// ---------------------------------------------------------------------------
// Persistent KalmanNet scan.  Workgroup b owns hidden indices [16b, 16b+16).
// ---------------------------------------------------------------------------
__global__ __launch_bounds__(NTHR, 1) void knet_persistent(
    const float* __restrict__ y, const float* __restrict__ f, const float* __restrict__ h,
    const float* __restrict__ m1x0,
    const float* __restrict__ b1, const float* __restrict__ b_ih,
    const float* __restrict__ b_hh, const float* __restrict__ b2,
    const float* __restrict__ b3,
    const __bf16* __restrict__ W1b, const __bf16* __restrict__ Wihb,
    const __bf16* __restrict__ Whhb, const __bf16* __restrict__ W2b,
    const __bf16* __restrict__ W3b,
    float* __restrict__ hnf, __bf16* __restrict__ hnb,        // [2][KHH]
    float* __restrict__ a2g, unsigned* __restrict__ bar,
    float* __restrict__ out, int T) {

  __shared__ __align__(16) __bf16 s_a1[H1];
  __shared__ float s_part[9][16];
  __shared__ float s_post[MDIM], s_prpv[MDIM], s_prior[MDIM];
  __shared__ float s_innov[NDIM], s_dx[MDIM], s_xin[MDIM + NDIM];
  __shared__ float s_kk[MDIM * NDIM];
  __shared__ float s_a2[H2];
  __shared__ float s_nrm[2];

  const int tid  = threadIdx.x;
  const int wave = tid >> 5;
  const int lane = tid & 31;
  const int bid  = blockIdx.x;
  const int i0   = 16 * bid;

  unsigned* cnt = bar + 0;
  unsigned* gen = bar + 1;

  if (tid < MDIM) { s_post[tid] = m1x0[tid]; s_prpv[tid] = m1x0[tid]; }

  for (int t = 0; t < T; ++t) {
    __syncthreads();

    // ---- stage 1 (redundant per WG): prior, innovation, features, a1 ----
    if (tid < MDIM) {
      float pr = 0.f;
      for (int j = 0; j < MDIM; ++j) pr += f[tid * MDIM + j] * s_post[j];
      s_prior[tid] = pr;
    }
    __syncthreads();
    if (tid < NDIM) {
      float my = 0.f;
      for (int j = 0; j < MDIM; ++j) my += h[tid * MDIM + j] * s_prior[j];
      s_innov[tid] = y[tid * T + t] - my;
    }
    if (tid < MDIM) s_dx[tid] = s_post[tid] - s_prpv[tid];
    __syncthreads();
    if (tid == 0) {
      float n1 = 0.f, n2 = 0.f;
      for (int j = 0; j < NDIM; ++j) n1 += s_innov[j] * s_innov[j];
      for (int j = 0; j < MDIM; ++j) n2 += s_dx[j] * s_dx[j];
      s_nrm[0] = fmaxf(sqrtf(n1), 1e-12f);
      s_nrm[1] = fmaxf(sqrtf(n2), 1e-12f);
    }
    __syncthreads();
    if (tid < MDIM + NDIM)
      s_xin[tid] = (tid < NDIM) ? s_innov[tid] / s_nrm[0] : s_dx[tid - NDIM] / s_nrm[1];
    __syncthreads();
    for (int r = tid; r < H1; r += NTHR) {
      const __bf16* wr = W1b + r * (MDIM + NDIM);
      float acc = b1[r];
      for (int j = 0; j < MDIM + NDIM; ++j) acc += (float)wr[j] * s_xin[j];
      s_a1[r] = (__bf16)fmaxf(acc, 0.f);
    }
    __syncthreads();

    // ---- phase A: gi/gh for this WG's 16 hidden indices + GRU update ----
    const int     buf    = t & 1;
    const float*  hf_cur = hnf + buf * KHH;
    float*        hf_nxt = hnf + (buf ^ 1) * KHH;
    const __bf16* hb_cur = hnb + buf * KHH;
    __bf16*       hb_nxt = hnb + (buf ^ 1) * KHH;

    {
      v8f acc = {0.f, 0.f, 0.f, 0.f, 0.f, 0.f, 0.f, 0.f};
      if (wave < 3) {                       // gate `wave`, full K over W_ih @ a1
        const __bf16* Wbase = Wihb + (size_t)(wave * HID + i0) * H1;
        acc = gemv_tile(Wbase, H1, s_a1, 0, H1 / 32, lane, acc);
      } else {                              // gate g, K-half hf over W_hh @ hn
        const int g  = (wave - 3) % 3;
        const int hf = (wave - 3) / 3;
        const __bf16* Wbase = Whhb + (size_t)(g * HID + i0) * KHH;
        acc = gemv_tile(Wbase, KHH, hb_cur, hf * (KHH / 2), KHH / 64, lane, acc);
      }
      if ((lane & 15) == 0) {
        const int half = lane >> 4;
        for (int r = 0; r < 8; ++r) s_part[wave][half * 8 + r] = acc[r];
      }
    }
    __syncthreads();
    if (tid < 16) {
      const int i = i0 + tid;
      const float gir = s_part[0][tid] + b_ih[i];
      const float giz = s_part[1][tid] + b_ih[HID + i];
      const float gin = s_part[2][tid] + b_ih[2 * HID + i];
      const float ghr = s_part[3][tid] + s_part[6][tid] + b_hh[i];
      const float ghz = s_part[4][tid] + s_part[7][tid] + b_hh[HID + i];
      const float ghn = s_part[5][tid] + s_part[8][tid] + b_hh[2 * HID + i];
      const float rg = 1.f / (1.f + expf(-(gir + ghr)));
      const float zg = 1.f / (1.f + expf(-(giz + ghz)));
      const float ng = tanhf(gin + rg * ghn);
      const float hnew = (1.f - zg) * ng + zg * hf_cur[i];
      hf_nxt[i] = hnew;
      hb_nxt[i] = (__bf16)hnew;
    }

    grid_barrier(cnt, gen);                 // hn_new complete

    // ---- phase B: a2 = relu(W2 @ hn_new + b2), WGs 0..24 ----
    if (bid < H2 / 16) {
      if (wave < 8) {
        v8f acc = {0.f, 0.f, 0.f, 0.f, 0.f, 0.f, 0.f, 0.f};
        const __bf16* Wbase = W2b + (size_t)(16 * bid) * KHH;
        acc = gemv_tile(Wbase, KHH, hb_nxt, wave * (KHH / 8), KHH / 256, lane, acc);
        if ((lane & 15) == 0) {
          const int half = lane >> 4;
          for (int r = 0; r < 8; ++r) s_part[wave][half * 8 + r] = acc[r];
        }
      }
      __syncthreads();
      if (tid < 16) {
        float s = b2[16 * bid + tid];
        for (int w = 0; w < 8; ++w) s += s_part[w][tid];
        a2g[16 * bid + tid] = fmaxf(s, 0.f);
      }
    }

    grid_barrier(cnt, gen);                 // a2 complete

    // ---- phase C (redundant per WG): K-gain, posterior update ----
    for (int j = tid; j < H2; j += NTHR) s_a2[j] = a2g[j];
    __syncthreads();
    if (tid < MDIM * NDIM) {
      const __bf16* wr = W3b + tid * H2;
      float acc = b3[tid];
      for (int j = 0; j < H2; ++j) acc += (float)wr[j] * s_a2[j];
      s_kk[tid] = acc;
    }
    __syncthreads();
    if (tid < MDIM) {
      float pn = s_prior[tid];
      for (int n = 0; n < NDIM; ++n) pn += s_kk[tid * NDIM + n] * s_innov[n];
      if (bid == 0) out[tid * T + t] = pn;  // xs.T: [M, T]
      s_prpv[tid] = s_prior[tid];
      s_post[tid] = pn;
    }
  }
}

// ---------------------------------------------------------------------------
extern "C" void kernel_launch(void* const* d_in, const int* in_sizes, int n_in,
                              void* d_out, int out_size, void* d_ws, size_t ws_size,
                              hipStream_t stream) {
  const float* y    = (const float*)d_in[0];
  const float* f    = (const float*)d_in[1];
  const float* h    = (const float*)d_in[2];
  const float* m1x0 = (const float*)d_in[3];
  const float* hn0  = (const float*)d_in[4];
  const float* W1   = (const float*)d_in[5];
  const float* b1   = (const float*)d_in[6];
  const float* W_ih = (const float*)d_in[7];
  const float* W_hh = (const float*)d_in[8];
  const float* b_ih = (const float*)d_in[9];
  const float* b_hh = (const float*)d_in[10];
  const float* W2   = (const float*)d_in[11];
  const float* b2   = (const float*)d_in[12];
  const float* W3   = (const float*)d_in[13];
  const float* b3   = (const float*)d_in[14];
  const int T = in_sizes[0] / NDIM;

  char* ws = (char*)d_ws;
  size_t off = 0;
  auto take = [&](size_t bytes) -> void* {
    void* p = ws + off;
    off = (off + bytes + 255) & ~(size_t)255;
    return p;
  };
  unsigned* bar  = (unsigned*)take(2 * sizeof(unsigned));
  float*    hnf  = (float*)   take(2 * KHH * sizeof(float));
  __bf16*   hnb  = (__bf16*)  take(2 * KHH * sizeof(__bf16));
  float*    a2g  = (float*)   take(H2 * sizeof(float));
  __bf16*   W1b  = (__bf16*)  take((size_t)H1 * (MDIM + NDIM) * sizeof(__bf16));
  __bf16*   W3b  = (__bf16*)  take((size_t)MDIM * NDIM * H2 * sizeof(__bf16));
  __bf16*   W2b  = (__bf16*)  take((size_t)H2 * KHH * sizeof(__bf16));
  __bf16*   Wihb = (__bf16*)  take((size_t)3 * HID * H1 * sizeof(__bf16));
  __bf16*   Whhb = (__bf16*)  take((size_t)3 * HID * KHH * sizeof(__bf16));

  k_cvt<<<64,   256, 0, stream>>>(W1,   W1b,  H1,          MDIM + NDIM, MDIM + NDIM);
  k_cvt<<<64,   256, 0, stream>>>(W3,   W3b,  MDIM * NDIM, H2,          H2);
  k_cvt<<<512,  256, 0, stream>>>(W2,   W2b,  H2,          HID,         KHH);
  k_cvt<<<2048, 256, 0, stream>>>(W_ih, Wihb, 3 * HID,     H1,          H1);
  k_cvt<<<2048, 256, 0, stream>>>(W_hh, Whhb, 3 * HID,     HID,         KHH);
  k_init<<<16,  256, 0, stream>>>(hnf, hnb, hn0, bar);

  knet_persistent<<<NWG, NTHR, 0, stream>>>(
      y, f, h, m1x0, b1, b_ih, b_hh, b2, b3,
      W1b, Wihb, Whhb, W2b, W3b,
      hnf, hnb, a2g, bar, (float*)d_out, T);
}
